// MoEW4A8OF16LinearDynamicInputScale_4861902979229
// MI455X (gfx1250) — 8-bit matrix operand order repaired and checked by reading, compiled, not run on hardware
//
#include <hip/hip_runtime.h>


typedef __attribute__((ext_vector_type(8))) int v8i;

typedef int vs2i __attribute__((vector_size(8)));
typedef int vs4i __attribute__((vector_size(16)));
typedef __attribute__((address_space(1))) vs2i* gas_b64_ptr;
typedef __attribute__((address_space(3))) vs2i* las_b64_ptr;
typedef __attribute__((address_space(1))) vs4i* gas_b128_ptr;
typedef __attribute__((address_space(3))) vs4i* las_b128_ptr;

#define NE 8
#define NT 1024
#define NK 2048
#define NO 5632
#define KSTEP 128
#define NKT (NK / KSTEP)
#define ABUF (32 * KSTEP)

__global__ __launch_bounds__(256) void
pack_x_int8_kernel(const int* __restrict__ x, unsigned* __restrict__ xp) {
    const int idx = blockIdx.x * 256 + threadIdx.x;
    const int4 v = ((const int4*)x)[idx];
    const unsigned p01 =
        __builtin_amdgcn_perm((unsigned)v.y, (unsigned)v.x, 0x0c0c0400u);
    const unsigned p23 =
        __builtin_amdgcn_perm((unsigned)v.w, (unsigned)v.z, 0x0c0c0400u);
    xp[idx] = __builtin_amdgcn_perm(p23, p01, 0x05040100u);
}

__global__ __launch_bounds__(256) void
MoEW4A8OF16LinearDynamicInputScale_4861902979229_kernel(
    const unsigned char* __restrict__ xp,
    const _Float16* __restrict__ input_scales,
    const _Float16* __restrict__ input_sum,
    const int* __restrict__ problem_sizes,
    const signed char* __restrict__ qweight,
    const _Float16* __restrict__ s1_scales,
    const _Float16* __restrict__ s1_szeros,
    const _Float16* __restrict__ bias,
    _Float16* __restrict__ out)
{
    __shared__ unsigned char lds_a[2 * ABUF];
    __shared__ _Float16 lds_xs[32];
    __shared__ _Float16 lds_xm[32];

    const int tid  = threadIdx.x;
    const int lane = tid & 31;
    const int wave = tid >> 5;

    int e = 0, t0 = 0, rows = 0, found = 0;
    {
        int y = blockIdx.y;
        int base = 0;
        for (int i = 0; i < NE; ++i) {
            const int psz = problem_sizes[i];
            const int ntile = (psz + 31) >> 5;
            if (!found) {
                if (y < ntile) {
                    e = i;
                    t0 = base + y * 32;
                    rows = psz - y * 32;
                    if (rows > 32) rows = 32;
                    found = 1;
                } else {
                    y -= ntile;
                }
            }
            base += psz;
        }
    }
    if (!found) return;

    const int n    = lane & 15;
    const int half = lane >> 4;
    const int o0   = blockIdx.x * 256 + wave * 32 + n;
    const int o1   = o0 + 16;

    const float sc0 = (float)s1_scales[e * NO + o0];
    const float sz0 = (float)s1_szeros[e * NO + o0];
    const float bi0 = (float)bias[e * NO + o0];
    const float sc1 = (float)s1_scales[e * NO + o1];
    const float sz1 = (float)s1_szeros[e * NO + o1];
    const float bi1 = (float)bias[e * NO + o1];

    const signed char* __restrict__ wp0 =
        qweight + ((size_t)e * NO + o0) * (size_t)NK;
    const signed char* __restrict__ wp1 =
        qweight + ((size_t)e * NO + o1) * (size_t)NK;

    if (tid < 32) {
        const int r = tid < rows ? tid : rows - 1;
        lds_xs[tid] = input_scales[t0 + r];
        lds_xm[tid] = input_sum[t0 + r];
    }

    const int srow = tid >> 3;
    const int soff = (tid & 7) * 16;
    const int srcr = srow < rows ? srow : rows - 1;
    const unsigned char* __restrict__ sp = xp + (size_t)(t0 + srcr) * NK + soff;

    auto stage = [&](int k0, int sb) {
        unsigned char* dst = &lds_a[sb + srow * KSTEP + soff];
#if __has_builtin(__builtin_amdgcn_global_load_async_to_lds_b128)
        __builtin_amdgcn_global_load_async_to_lds_b128(
            (gas_b128_ptr)(sp + k0), (las_b128_ptr)dst, 0, 0);
#elif __has_builtin(__builtin_amdgcn_global_load_async_to_lds_b64)
        __builtin_amdgcn_global_load_async_to_lds_b64(
            (gas_b64_ptr)(sp + k0), (las_b64_ptr)dst, 0, 0);
        __builtin_amdgcn_global_load_async_to_lds_b64(
            (gas_b64_ptr)(sp + k0 + 8), (las_b64_ptr)(dst + 8), 0, 0);
#else
        const int4 v = *(const int4*)(sp + k0);
        *(int4*)dst = v;
#endif
    };

    auto wait_stage = [&]() {
#if __has_builtin(__builtin_amdgcn_global_load_async_to_lds_b64)
#if __has_builtin(__builtin_amdgcn_s_wait_asynccnt)
        __builtin_amdgcn_s_wait_asynccnt(0);
#else
        asm volatile("s_wait_asynccnt 0x0" ::: "memory");
#endif
#endif
        __syncthreads();
    };

    v8i acc00 = {};
    v8i acc10 = {};
    v8i acc01 = {};
    v8i acc11 = {};

    const int abase0 = n * KSTEP + half * 16;
    const int abase1 = (n + 16) * KSTEP + half * 16;

    auto compute = [&](int k0, int lb) {
#pragma unroll
        for (int ks = 0; ks < KSTEP; ks += 64) {
            const unsigned char* la = &lds_a[lb + ks];
            const int2 a01 = *(const int2*)(la + abase0 + 0);
            const int2 a23 = *(const int2*)(la + abase0 + 8);
            const int2 a45 = *(const int2*)(la + abase0 + 32);
            const int2 a67 = *(const int2*)(la + abase0 + 40);
            const v8i a_lo = {a01.x, a01.y, a23.x, a23.y,
                              a45.x, a45.y, a67.x, a67.y};

            const int2 c01 = *(const int2*)(la + abase1 + 0);
            const int2 c23 = *(const int2*)(la + abase1 + 8);
            const int2 c45 = *(const int2*)(la + abase1 + 32);
            const int2 c67 = *(const int2*)(la + abase1 + 40);
            const v8i a_hi = {c01.x, c01.y, c23.x, c23.y,
                              c45.x, c45.y, c67.x, c67.y};

            const int4 p03 = *(const int4*)(wp0 + k0 + ks + half * 16);
            const int4 p47 = *(const int4*)(wp0 + k0 + ks + 32 + half * 16);
            const v8i b0 = {p03.x, p03.y, p03.z, p03.w,
                            p47.x, p47.y, p47.z, p47.w};

            const int4 q03 = *(const int4*)(wp1 + k0 + ks + half * 16);
            const int4 q47 = *(const int4*)(wp1 + k0 + ks + 32 + half * 16);
            const v8i b1 = {q03.x, q03.y, q03.z, q03.w,
                            q47.x, q47.y, q47.z, q47.w};

            acc00 = __builtin_amdgcn_wmma_i32_16x16x64_iu8(true, a_lo, false, b0,
                                                           acc00, false, false);
            acc10 = __builtin_amdgcn_wmma_i32_16x16x64_iu8(true, a_hi, false, b0,
                                                           acc10, false, false);
            acc01 = __builtin_amdgcn_wmma_i32_16x16x64_iu8(true, a_lo, false, b1,
                                                           acc01, false, false);
            acc11 = __builtin_amdgcn_wmma_i32_16x16x64_iu8(true, a_hi, false, b1,
                                                           acc11, false, false);
        }
        __builtin_prefetch(wp0 + k0 + 2 * KSTEP, 0, 3);
        __builtin_prefetch(wp1 + k0 + 2 * KSTEP, 0, 3);
    };

    stage(0, 0);
    wait_stage();

    int loff = 0;
#pragma unroll 1
    for (int kt = 0; kt < NKT - 1; ++kt) {
        stage((kt + 1) * KSTEP, loff ^ ABUF);
        compute(kt * KSTEP, loff);
        wait_stage();
        loff ^= ABUF;
    }
    compute((NKT - 1) * KSTEP, loff);

#pragma unroll
    for (int r = 0; r < 8; ++r) {
        const int mm0 = r + half * 8;
        const int mm1 = mm0 + 16;
        if (mm0 < rows) {
            const float xs = (float)lds_xs[mm0];
            const float xm = (float)lds_xm[mm0];
            const size_t rowoff = (size_t)(t0 + mm0) * NO;
            __builtin_nontemporal_store(
                (_Float16)(xs * (sc0 * (float)acc00[r] - xm * sz0) + bi0),
                out + rowoff + o0);
            __builtin_nontemporal_store(
                (_Float16)(xs * (sc1 * (float)acc01[r] - xm * sz1) + bi1),
                out + rowoff + o1);
        }
        if (mm1 < rows) {
            const float xs = (float)lds_xs[mm1];
            const float xm = (float)lds_xm[mm1];
            const size_t rowoff = (size_t)(t0 + mm1) * NO;
            __builtin_nontemporal_store(
                (_Float16)(xs * (sc0 * (float)acc10[r] - xm * sz0) + bi0),
                out + rowoff + o0);
            __builtin_nontemporal_store(
                (_Float16)(xs * (sc1 * (float)acc11[r] - xm * sz1) + bi1),
                out + rowoff + o1);
        }
    }
}

extern "C" void kernel_launch(void* const* d_in, const int* in_sizes, int n_in,
                              void* d_out, int out_size, void* d_ws, size_t ws_size,
                              hipStream_t stream) {
    const int*         x    = (const int*)d_in[0];
    const _Float16*    isc  = (const _Float16*)d_in[1];
    const _Float16*    isum = (const _Float16*)d_in[2];
    const int*         psz  = (const int*)d_in[3];
    const signed char* qw   = (const signed char*)d_in[4];
    const _Float16*    s1s  = (const _Float16*)d_in[5];
    const _Float16*    s1z  = (const _Float16*)d_in[6];
    const _Float16*    bs   = (const _Float16*)d_in[7];
    _Float16*          outp = (_Float16*)d_out;

    (void)in_sizes; (void)n_in; (void)out_size; (void)ws_size;

    unsigned* xpack = (unsigned*)d_ws;
    pack_x_int8_kernel<<<dim3(NT * NK / 1024), dim3(256), 0, stream>>>(x, xpack);

    dim3 grid(NO / 256, NT / 32 + NE, 1);
    MoEW4A8OF16LinearDynamicInputScale_4861902979229_kernel<<<grid, dim3(256), 0,
                                                              stream>>>(
        (const unsigned char*)xpack, isc, isum, psz, qw, s1s, s1z, bs, outp);
}
